// BDH_21191368638898
// MI455X (gfx1250) — matrix operand order repaired and checked by reading, not compiled, not run on hardware
//
#include <hip/hip_runtime.h>
#include <hip/hip_bf16.h>

typedef __bf16 bf16;
typedef __attribute__((ext_vector_type(16))) __bf16 v16bf;
typedef __attribute__((ext_vector_type(8)))  __bf16 v8bf;
typedef __attribute__((ext_vector_type(8)))  float  v8f;

#define DEV static __device__ __forceinline__

#define TT    1024   // tokens
#define DD    256    // model dim
#define NHEAD 4
#define NN    2048   // per-head key dim
#define COLS  8192   // NHEAD*NN
#define NCH   16     // chunks
#define CHK   64     // chunk length
#define LDSA_LD 264  // 256 + 8 pad: 528B row stride -> conflict-free frag reads

DEV bf16 f2b(float f) {
  unsigned u = __builtin_bit_cast(unsigned, f);
  u += 0x7FFFu + ((u >> 16) & 1u);
  unsigned short h = (unsigned short)(u >> 16);
  return __builtin_bit_cast(bf16, h);
}
DEV float b2f(bf16 b) {
  unsigned short h = __builtin_bit_cast(unsigned short, b);
  unsigned u = ((unsigned)h) << 16;
  return __builtin_bit_cast(float, u);
}
DEV v8f v8f_zero() {
  v8f z;
#pragma unroll
  for (int i = 0; i < 8; ++i) z[i] = 0.f;
  return z;
}
DEV float sqrelu_f(float v) { float r = v > 0.f ? v : 0.f; return r * r; }

// A fragment: 16x32 bf16 tile from row-major [M][K] (ld in elements).
// lanes 0-15: row=lane, K=0..7 & 16..23 ; lanes 16-31: row=lane-16, K=8..15 & 24..31
DEV v16bf load_frag_a(const bf16* tile, int ld) {
  int lane = threadIdx.x & 31;
  const bf16* p = tile + (size_t)(lane & 15) * ld + ((lane >> 4) << 3);
  v8bf lo = *(const v8bf*)p;
  v8bf hi = *(const v8bf*)(p + 16);
  v16bf r;
#pragma unroll
  for (int i = 0; i < 8; ++i) { r[i] = lo[i]; r[i + 8] = hi[i]; }
  return r;
}
// B fragment: 32x16 bf16 (K x N) sourced from W[N][K] row-major.
// lanes 0-15: n=lane, K=0..7 & 16..23 ; lanes 16-31: n=lane-16, K=8..15 & 24..31
DEV v16bf load_frag_b(const bf16* tile, int ld) {
  int lane = threadIdx.x & 31;
  const bf16* p = tile + (size_t)(lane & 15) * ld + ((lane >> 4) << 3);
  v8bf lo = *(const v8bf*)p;
  v8bf hi = *(const v8bf*)(p + 16);
  v16bf r;
#pragma unroll
  for (int i = 0; i < 8; ++i) { r[i] = lo[i]; r[i + 8] = hi[i]; }
  return r;
}
DEV v8f wmma_bf(v16bf a, v16bf b, v8f c) {
  return __builtin_amdgcn_wmma_f32_16x16x32_bf16(false, a, false, b, (short)0, c,
                                                 false, false);
}

// 16B async global->LDS copy (gfx1250). lds_addr = low 32 bits of generic
// pointer to __shared__ (== wave-relative LDS offset per aperture rules).
DEV void async_g2l_b128(unsigned lds_addr, const bf16* gaddr) {
  asm volatile("global_load_async_to_lds_b128 %0, %1, off"
               :: "v"(lds_addr), "v"(gaddr)
               : "memory");
}
DEV void wait_asynccnt0() {
  asm volatile("s_wait_asynccnt 0x0" ::: "memory");
}

DEV float block_sum(float v) {
#pragma unroll
  for (int o = 16; o > 0; o >>= 1) v += __shfl_xor(v, o, 32);
  __shared__ float red[8];
  int w = threadIdx.x >> 5, lane = threadIdx.x & 31;
  if (lane == 0) red[w] = v;
  __syncthreads();
  float s = (lane < 8) ? red[lane] : 0.f;
#pragma unroll
  for (int o = 4; o > 0; o >>= 1) s += __shfl_xor(s, o, 32);
  s = __shfl(s, 0, 32);
  __syncthreads();
  return s;
}

// ---------------- elementwise / reduction kernels ----------------

__global__ void cvt_bf16_kernel(const float* __restrict__ src, bf16* __restrict__ dst, long n) {
  long i = (long)blockIdx.x * 256 + threadIdx.x;
  if (i < n) dst[i] = f2b(src[i]);
}

__global__ void zero_bf16_kernel(bf16* __restrict__ p, long n) {
  long i = (long)blockIdx.x * 256 + threadIdx.x;
  if (i < n) p[i] = f2b(0.f);
}

__global__ void embed_rms_kernel(const float* __restrict__ embed_w, const int* __restrict__ idx,
                                 float* __restrict__ x, float* __restrict__ x0) {
  int t = blockIdx.x, d = threadIdx.x;
  float v = embed_w[(size_t)idx[t] * DD + d];
  float ss = block_sum(v * v);
  float r = v * rsqrtf(ss / (float)DD + 1e-5f);
  x[(size_t)t * DD + d]  = r;
  x0[(size_t)t * DD + d] = r;
}

__global__ void resid_mix_kernel(const float* __restrict__ x, const float* __restrict__ x0,
                                 const float* __restrict__ rl, const float* __restrict__ xl,
                                 int layer, float* __restrict__ xin,
                                 bf16* __restrict__ xinb, bf16* __restrict__ xinT) {
  int gid = blockIdx.x * 256 + threadIdx.x;
  int t = gid >> 8, d = gid & 255;
  float v = rl[layer] * x[gid] + xl[layer] * x0[gid];
  xin[gid] = v;
  bf16 b = f2b(v);
  xinb[gid] = b;
  xinT[(size_t)d * TT + t] = b;  // v^T for GLA
}

__global__ void rope_kernel(const float* __restrict__ xs, bf16* __restrict__ q) {
  int gid = blockIdx.x * 256 + threadIdx.x;  // 1024*32*128 pairs
  int i  = gid & 127;
  int cc = (gid >> 7) & 31;
  int t  = gid >> 12;
  size_t base = (size_t)t * COLS + (size_t)cc * 256;
  float x1 = xs[base + i], x2 = xs[base + 128 + i];
  float inv_freq = exp2f(-18.f * (float)(2 * i) / 256.f);  // ROPE_BASE=2^18, CS=256
  float fr = (float)t * inv_freq;
  float xsc = ((float)(2 * i) + 0.4f * 256.f) / (1.4f * 256.f);
  float power = ((float)t - 512.f) / 512.f;  // SCALE_BASE=512
  float sc = __powf(xsc, power);
  float cv = __cosf(fr) * sc, sv = __sinf(fr) * sc;
  q[base + i]       = f2b(x1 * cv - x2 * sv);
  q[base + 128 + i] = f2b(x2 * cv + x1 * sv);
}

// per (chunk c, column j): cumsum of g=-gate, emit qg, kg, kS^T, exp(g_last)
__global__ void gla_pre_kernel(const bf16* __restrict__ qb, const float* __restrict__ gate,
                               bf16* __restrict__ qg, bf16* __restrict__ kg,
                               bf16* __restrict__ kST, float* __restrict__ egl) {
  int j = blockIdx.x * 256 + threadIdx.x;  // 0..8191
  int c = blockIdx.y;                      // 0..15
  float gl = 0.f;
  for (int t = 0; t < CHK; ++t) gl -= gate[(size_t)(c * CHK + t) * COLS + j];
  egl[(size_t)c * COLS + j] = __expf(gl);
  const float scale = rsqrtf((float)NN);
  float acc = 0.f;
  for (int t = 0; t < CHK; ++t) {
    size_t idx = (size_t)(c * CHK + t) * COLS + j;
    acc -= gate[idx];
    float qv = b2f(qb[idx]);
    qg[idx] = f2b(qv * __expf(acc) * scale);
    kg[idx] = f2b(qv * __expf(-acc));
    kST[(size_t)j * TT + (c * CHK + t)] = f2b(qv * __expf(gl - acc));
  }
}

__global__ void layernorm_o_kernel(const float* __restrict__ o, bf16* __restrict__ oln) {
  int b = blockIdx.x;  // t*NHEAD + h
  int d = threadIdx.x;
  float v = o[(size_t)b * DD + d];
  float s1 = block_sum(v);
  float s2 = block_sum(v * v);
  float m = s1 / (float)DD;
  float var = s2 / (float)DD - m * m;
  oln[(size_t)b * DD + d] = f2b((v - m) * rsqrtf(var + 1e-5f));
}

__global__ void post_dec_kernel(const float* __restrict__ y, const float* __restrict__ xin,
                                float* __restrict__ x) {
  int t = blockIdx.x, d = threadIdx.x;
  float v = y[(size_t)t * DD + d];
  float s1 = block_sum(v);
  float s2 = block_sum(v * v);
  float m = s1 / (float)DD, var = s2 / (float)DD - m * m;
  float z = (v - m) * rsqrtf(var + 1e-5f) + xin[(size_t)t * DD + d];
  float s3 = block_sum(z * z);
  x[(size_t)t * DD + d] = z * rsqrtf(s3 / (float)DD + 1e-5f);
}

__global__ void backout_kernel(const float* __restrict__ x, const float* __restrict__ x0,
                               const float* __restrict__ bl, bf16* __restrict__ xf) {
  int t = blockIdx.x, d = threadIdx.x;
  float z = x[(size_t)t * DD + d] - bl[0] * x0[(size_t)t * DD + d];
  float s = block_sum(z * z);
  xf[(size_t)t * DD + d] = f2b(z * rsqrtf(s / (float)DD + 1e-5f));
}

// ---------------- generic bf16 WMMA GEMM: out = epi(A[MxK] . W[NxK]^T) ----------------
// 8 waves, wave tile 32x32, block tile 64x128. epi: 0=sqrelu*post (f32),
// 1=bf16(xs*sqrelu) (f32 xs at same [row][colOff+col]), 2=raw f32.
// use_lds_a=1 (requires K==256): stage the 64xK A panel in LDS once per block
// via global_load_async_to_lds_b128 (ASYNCcnt path), then serve all 8 waves'
// A fragments from LDS -> removes 4x redundant global A traffic.
__global__ __launch_bounds__(256) void gemm_kernel(
    const bf16* __restrict__ A, long lda, long a_zoff,
    const bf16* __restrict__ W, long ldw, long w_zoff,
    float* __restrict__ outF, bf16* __restrict__ outB, long ldo, long o_zcol,
    const float* __restrict__ xsmul, int K, float post, int epi, int use_lds_a) {
  __shared__ __align__(32) bf16 Asmem[64 * LDSA_LD];
  int z = blockIdx.z;
  const bf16* Az = A + (size_t)a_zoff * z;
  const bf16* Wz = W + (size_t)w_zoff * z;
  long colOff = o_zcol * (long)z;
  int w = threadIdx.x >> 5, lane = threadIdx.x & 31;
  int m0 = blockIdx.x * 64 + (w >> 2) * 32;
  int mloc = (w >> 2) * 32;
  int n0 = blockIdx.y * 128 + (w & 3) * 32;
  v8f c[2][2];
#pragma unroll
  for (int i = 0; i < 2; ++i)
#pragma unroll
    for (int j = 0; j < 2; ++j) c[i][j] = v8f_zero();

  if (use_lds_a) {
    // 64 rows x K cols, 16B chunks: K/8 chunks per row, 256 threads cooperate.
    int cpr = K >> 3;  // chunks per row (32 for K=256)
    for (int qc = threadIdx.x; qc < 64 * cpr; qc += 256) {
      int row = qc / cpr;
      int kc  = qc - row * cpr;
      const bf16* g = Az + (size_t)(blockIdx.x * 64 + row) * lda + kc * 8;
      unsigned l = (unsigned)(size_t)&Asmem[row * LDSA_LD + kc * 8];
      async_g2l_b128(l, g);
    }
    wait_asynccnt0();
    __syncthreads();
    for (int k0 = 0; k0 < K; k0 += 32) {
      v16bf a0 = load_frag_a(&Asmem[(size_t)mloc * LDSA_LD + k0], LDSA_LD);
      v16bf a1 = load_frag_a(&Asmem[(size_t)(mloc + 16) * LDSA_LD + k0], LDSA_LD);
      v16bf b0 = load_frag_b(Wz + (size_t)n0 * ldw + k0, (int)ldw);
      v16bf b1 = load_frag_b(Wz + (size_t)(n0 + 16) * ldw + k0, (int)ldw);
      c[0][0] = wmma_bf(a0, b0, c[0][0]);
      c[0][1] = wmma_bf(a0, b1, c[0][1]);
      c[1][0] = wmma_bf(a1, b0, c[1][0]);
      c[1][1] = wmma_bf(a1, b1, c[1][1]);
    }
  } else {
    for (int k0 = 0; k0 < K; k0 += 32) {
      v16bf a0 = load_frag_a(Az + (size_t)m0 * lda + k0, (int)lda);
      v16bf a1 = load_frag_a(Az + (size_t)(m0 + 16) * lda + k0, (int)lda);
      v16bf b0 = load_frag_b(Wz + (size_t)n0 * ldw + k0, (int)ldw);
      v16bf b1 = load_frag_b(Wz + (size_t)(n0 + 16) * ldw + k0, (int)ldw);
      c[0][0] = wmma_bf(a0, b0, c[0][0]);
      c[0][1] = wmma_bf(a0, b1, c[0][1]);
      c[1][0] = wmma_bf(a1, b0, c[1][0]);
      c[1][1] = wmma_bf(a1, b1, c[1][1]);
    }
  }

  int rbase = (lane >> 4) << 3;
  int coll = lane & 15;
#pragma unroll
  for (int i = 0; i < 2; ++i)
#pragma unroll
    for (int j = 0; j < 2; ++j)
#pragma unroll
      for (int r = 0; r < 8; ++r) {
        long row = m0 + 16 * i + rbase + r;
        long col = colOff + n0 + 16 * j + coll;
        size_t oi = (size_t)row * ldo + col;
        float v = c[i][j][r];
        if (epi == 0)      outF[oi] = sqrelu_f(v) * post;
        else if (epi == 1) outB[oi] = f2b(xsmul[oi] * sqrelu_f(v));
        else               outF[oi] = v;
      }
}

// ---------------- chunked GLA scan (one block per head x 64-wide Dv block) ----------------
__global__ __launch_bounds__(256) void gla_kernel(
    const bf16* __restrict__ qg, const bf16* __restrict__ kg,
    const bf16* __restrict__ kST, const bf16* __restrict__ xinT,
    const float* __restrict__ egl, bf16* __restrict__ St, float* __restrict__ o) {
  int head = blockIdx.y;
  int dvb  = blockIdx.x;  // 64 dv columns each
  int w = threadIdx.x >> 5, lane = threadIdx.x & 31;
  int rbase = (lane >> 4) << 3, coll = lane & 15;
  __shared__ __align__(32) bf16 As[64 * 80];

  const bf16* qgh  = qg + head * NN;                  // [t][n], ld COLS
  const bf16* kgh  = kg + head * NN;
  const bf16* kSTh = kST + (size_t)head * NN * TT;    // [n][t], ld TT
  bf16* Sth = St + (size_t)head * DD * NN;            // S^T: [d][n], ld NN

  for (int c = 0; c < NCH; ++c) {
    int t0 = c * CHK;
    v8f acc[4];
#pragma unroll
    for (int j = 0; j < 4; ++j) acc[j] = v8f_zero();

    if (w < 4) {
      // A[16w..16w+16][0..64] = qg . kg^T over K=NN
      const bf16* arow = qgh + (size_t)(t0 + 16 * w) * COLS;
      for (int k0 = 0; k0 < NN; k0 += 32) {
        v16bf af = load_frag_a(arow + k0, COLS);
#pragma unroll
        for (int j = 0; j < 4; ++j) {
          v16bf bfr = load_frag_b(kgh + (size_t)(t0 + 16 * j) * COLS + k0, COLS);
          acc[j] = wmma_bf(af, bfr, acc[j]);
        }
      }
#pragma unroll
      for (int j = 0; j < 4; ++j)
#pragma unroll
        for (int r = 0; r < 8; ++r) {
          int row = 16 * w + rbase + r;
          int col = 16 * j + coll;
          As[row * 80 + col] = f2b(row >= col ? acc[j][r] : 0.f);
        }
    } else {
      // o_inter[16(w-4)..][dv 64] = qg . S^T over K=NN (reads previous-chunk S)
      const bf16* arow = qgh + (size_t)(t0 + 16 * (w - 4)) * COLS;
      for (int k0 = 0; k0 < NN; k0 += 32) {
        v16bf af = load_frag_a(arow + k0, COLS);
#pragma unroll
        for (int j = 0; j < 4; ++j) {
          v16bf bfr = load_frag_b(Sth + (size_t)(dvb * 64 + 16 * j) * NN + k0, NN);
          acc[j] = wmma_bf(af, bfr, acc[j]);
        }
      }
    }
    __syncthreads();

    if (w < 4) {
      // state update: S^T[d][n] = S^T*exp(g_last)[n] + (v^T . kS)[d][n]; n-strip per wave
      const float* egh = egl + (size_t)c * COLS + head * NN;
      for (int nt = 0; nt < 32; ++nt) {
        int n0 = 512 * w + nt * 16;
        v8f s[4];
#pragma unroll
        for (int i = 0; i < 4; ++i) s[i] = v8f_zero();
        for (int k0 = 0; k0 < CHK; k0 += 32) {
          v16bf bfr = load_frag_b(kSTh + (size_t)n0 * TT + t0 + k0, TT);
#pragma unroll
          for (int i = 0; i < 4; ++i) {
            v16bf af = load_frag_a(xinT + (size_t)(dvb * 64 + 16 * i) * TT + t0 + k0, TT);
            s[i] = wmma_bf(af, bfr, s[i]);
          }
        }
#pragma unroll
        for (int i = 0; i < 4; ++i)
#pragma unroll
          for (int r = 0; r < 8; ++r) {
            int drow = dvb * 64 + 16 * i + rbase + r;
            int ncol = n0 + coll;
            size_t si = (size_t)drow * NN + ncol;
            Sth[si] = f2b(b2f(Sth[si]) * egh[ncol] + s[i][r]);
          }
      }
    } else {
      // o_intra: (A∘mask).v from LDS, K=CHK; then write o
      int tw = 16 * (w - 4);
      for (int k0 = 0; k0 < CHK; k0 += 32) {
        v16bf af = load_frag_a(&As[tw * 80 + k0], 80);
#pragma unroll
        for (int j = 0; j < 4; ++j) {
          v16bf bfr = load_frag_b(xinT + (size_t)(dvb * 64 + 16 * j) * TT + t0 + k0, TT);
          acc[j] = wmma_bf(af, bfr, acc[j]);
        }
      }
#pragma unroll
      for (int j = 0; j < 4; ++j)
#pragma unroll
        for (int r = 0; r < 8; ++r) {
          int trow = t0 + tw + rbase + r;
          int dcol = dvb * 64 + 16 * j + coll;
          o[((size_t)trow * NHEAD + head) * DD + dcol] = acc[j][r];
        }
    }
    __syncthreads();
  }
}

// ---------------- host orchestration ----------------

extern "C" void kernel_launch(void* const* d_in, const int* in_sizes, int n_in,
                              void* d_out, int out_size, void* d_ws, size_t ws_size,
                              hipStream_t stream) {
  (void)in_sizes; (void)n_in; (void)out_size; (void)ws_size;
  const float* embed_w  = (const float*)d_in[0];
  const float* lm_head  = (const float*)d_in[1];
  const float* enc_w    = (const float*)d_in[2];
  const float* enc_g_w  = (const float*)d_in[3];
  const float* dec_w    = (const float*)d_in[4];
  const float* enc_v_w  = (const float*)d_in[5];
  const float* b_lambda = (const float*)d_in[6];
  const float* r_lambda = (const float*)d_in[7];
  const float* x_lambda = (const float*)d_in[8];
  const int*   idx      = (const int*)d_in[9];
  float* logits = (float*)d_out;

  char* ws = (char*)d_ws;
  size_t off = 0;
  auto alloc = [&](size_t bytes) -> void* {
    void* p = ws + off;
    off += (bytes + 255) & ~(size_t)255;
    return p;
  };
  bf16* wEnc  = (bf16*)alloc(8192ull * 256 * 2);
  bf16* wGate = (bf16*)alloc(8192ull * 256 * 2);
  bf16* wDec  = (bf16*)alloc(256ull * 8192 * 2);
  bf16* wEncV = (bf16*)alloc(4ull * 2048 * 256 * 2);
  bf16* wHead = (bf16*)alloc(50304ull * 256 * 2);
  float* x    = (float*)alloc(1024ull * 256 * 4);
  float* x0   = (float*)alloc(1024ull * 256 * 4);
  float* xin  = (float*)alloc(1024ull * 256 * 4);
  bf16* xinb  = (bf16*)alloc(1024ull * 256 * 2);
  bf16* xinT  = (bf16*)alloc(256ull * 1024 * 2);
  float* xs   = (float*)alloc(1024ull * 8192 * 4);
  float* gate = (float*)alloc(1024ull * 8192 * 4);
  bf16* qb    = (bf16*)alloc(1024ull * 8192 * 2);  // q (rope), becomes qg in place
  bf16* kg    = (bf16*)alloc(1024ull * 8192 * 2);
  bf16* kST   = (bf16*)alloc(8192ull * 1024 * 2);
  float* egl  = (float*)alloc(16ull * 8192 * 4);
  bf16* Stw   = (bf16*)alloc(4ull * 256 * 2048 * 2);
  float* ob   = (float*)alloc(1024ull * 4 * 256 * 4);
  bf16* oln   = (bf16*)alloc(1024ull * 4 * 256 * 2);
  bf16* xy    = (bf16*)alloc(1024ull * 8192 * 2);
  float* yb   = (float*)alloc(1024ull * 256 * 4);
  bf16* xf    = (bf16*)alloc(1024ull * 256 * 2);

  auto cvt = [&](const float* s, bf16* d, long n) {
    cvt_bf16_kernel<<<dim3((unsigned)((n + 255) / 256)), 256, 0, stream>>>(s, d, n);
  };
  cvt(enc_w,   wEnc,  8192l * 256);
  cvt(enc_g_w, wGate, 8192l * 256);
  cvt(dec_w,   wDec,  256l * 8192);
  cvt(enc_v_w, wEncV, 4l * 2048 * 256);
  cvt(lm_head, wHead, 50304l * 256);

  embed_rms_kernel<<<1024, 256, 0, stream>>>(embed_w, idx, x, x0);

  for (int l = 0; l < 4; ++l) {
    resid_mix_kernel<<<1024, 256, 0, stream>>>(x, x0, r_lambda, x_lambda, l, xin, xinb, xinT);
    // xs = sqrelu(xin @ enc_w^T)   (K=256 -> async LDS A staging)
    gemm_kernel<<<dim3(16, 64, 1), 256, 0, stream>>>(
        xinb, 256, 0, wEnc, 256, 0, xs, nullptr, 8192, 0, nullptr, 256, 1.0f, 0, 1);
    // gate = sqrelu(xin @ enc_gate_w^T) / 1024
    gemm_kernel<<<dim3(16, 64, 1), 256, 0, stream>>>(
        xinb, 256, 0, wGate, 256, 0, gate, nullptr, 8192, 0, nullptr, 256, 1.0f / 1024.0f, 0, 1);
    rope_kernel<<<dim3(16384), 256, 0, stream>>>(xs, qb);
    gla_pre_kernel<<<dim3(32, 16), 256, 0, stream>>>(qb, gate, qb, kg, kST, egl);
    zero_bf16_kernel<<<dim3((unsigned)((4l * 256 * 2048 + 255) / 256)), 256, 0, stream>>>(
        Stw, 4l * 256 * 2048);
    gla_kernel<<<dim3(4, 4), 256, 0, stream>>>(qb, kg, kST, xinT, egl, Stw, ob);
    layernorm_o_kernel<<<4096, 256, 0, stream>>>(ob, oln);
    // xy = xs * sqrelu(o_ln @ enc_v^T)  (per head via grid z; K=256 -> LDS A)
    gemm_kernel<<<dim3(16, 16, 4), 256, 0, stream>>>(
        oln, 1024, 256, wEncV, 256, 2048l * 256, nullptr, xy, 8192, 2048, xs, 256, 1.0f, 1, 1);
    // y = xy @ dec_w^T   (K=8192 -> streaming, no LDS staging)
    gemm_kernel<<<dim3(16, 2, 1), 256, 0, stream>>>(
        xy, 8192, 0, wDec, 8192, 0, yb, nullptr, 256, 0, nullptr, 8192, 1.0f, 2, 0);
    post_dec_kernel<<<1024, 256, 0, stream>>>(yb, xin, x);
  }
  backout_kernel<<<1024, 256, 0, stream>>>(x, x0, b_lambda, xf);
  // logits = xf @ lm_head^T   (50304 = 128*393; K=256 -> LDS A panel reused)
  gemm_kernel<<<dim3(16, 393, 1), 256, 0, stream>>>(
      xf, 256, 0, wHead, 256, 0, logits, nullptr, 50304, 0, nullptr, 256, 1.0f, 2, 1);
}
